// NTM_24970939859202
// MI455X (gfx1250) — compile-verified, matrix operand order checked by reading, not run on hardware
//
#include <hip/hip_runtime.h>
#include <hip/hip_bf16.h>

// MI455X (gfx1250) fused NTM kernel.
// Memory floor: ~103MB @ 23.3TB/s ~= 4.4us. GEMM (1.7 GFLOP) runs on
// v_wmma_f32_16x16x32_bf16 (fp32 accum), everything else fused in-register /
// in-LDS so the only global store is N output floats.

typedef __bf16 bf16_t;
typedef __attribute__((ext_vector_type(16))) __bf16 v16bf;
typedef __attribute__((ext_vector_type(8)))  float  v8f;

#define D_    64
#define K_    100
#define KPAD  112      // 7 N-tiles of 16
#define C_    5
#define WAVES 8
#define SPW   16       // samples per wave (WMMA M)
#define SPB   128      // samples per block

__global__ __launch_bounds__(256)
void ntm_fused_kernel(const int* __restrict__ iI, const int* __restrict__ jI,
                      const int* __restrict__ kI,
                      const float* __restrict__ uE, const float* __restrict__ itE,
                      const float* __restrict__ tE,
                      const float* __restrict__ w,
                      const float* __restrict__ U1, const float* __restrict__ U2,
                      const float* __restrict__ U3,
                      const float* __restrict__ fcw, const float* __restrict__ fcb,
                      float* __restrict__ out, int Nn)
{
    __shared__ bf16_t wT[KPAD][D_];          // 14336 B  w^T, bf16, zero-padded cols
    __shared__ float  fcg[KPAD];             //   448 B  fc weights for gcp part
    __shared__ float  fct[C_*C_*C_];         //   500 B  fc weights for tucker part
    __shared__ float  Us[3][D_*C_];          //  3840 B  U1,U2,U3
    __shared__ float  xs[WAVES][SPW][D_];    // 32768 B  ie*je*ke staged per wave
    __shared__ float  abc[WAVES][SPW][3*C_]; //  7680 B  per-sample a,b,c projections
    __shared__ float  gpart[WAVES][SPW];     //   512 B  gcp branch contribution

    const int tid  = threadIdx.x;
    const int wv   = tid >> 5;
    const int ln   = tid & 31;
    const int base = blockIdx.x * SPB;

    // ---------------- init block-shared constants ----------------
    for (int idx = tid; idx < KPAD * D_; idx += 256) {
        int n = idx >> 6, d = idx & 63;
        wT[n][d] = (n < K_) ? (bf16_t)w[d * K_ + n] : (bf16_t)0.0f;
    }
    if (tid < KPAD)      fcg[tid] = (tid < K_) ? fcw[tid] : 0.0f;
    if (tid < C_*C_*C_)  fct[tid] = fcw[K_ + tid];
    for (int idx = tid; idx < 3 * D_ * C_; idx += 256) {
        int u = idx / (D_ * C_), r = idx % (D_ * C_);
        const float* Up = (u == 0) ? U1 : ((u == 1) ? U2 : U3);
        Us[u][r] = Up[r];
    }
    __syncthreads();

    // ---------------- phase 1: coalesced gathers + x product + projections ----
    const int sbase = base + wv * SPW;
    for (int s = 0; s < SPW; ++s) {
        int sidx = sbase + s; if (sidx >= Nn) sidx = Nn - 1;   // clamp (N%128==0 anyway)
        const int gi = iI[sidx], gj = jI[sidx], gk = kI[sidx];
        const float2 e1 = *(const float2*)(uE  + (size_t)gi * D_ + ln * 2);
        const float2 e2 = *(const float2*)(itE + (size_t)gj * D_ + ln * 2);
        const float2 e3 = *(const float2*)(tE  + (size_t)gk * D_ + ln * 2);
        xs[wv][s][ln * 2 + 0] = e1.x * e2.x * e3.x;
        xs[wv][s][ln * 2 + 1] = e1.y * e2.y * e3.y;

        // a = ie@U1, b = je@U2, c = ke@U3  (each lane owns 2 of the 64 d's)
        float pr[3 * C_];
        #pragma unroll
        for (int j = 0; j < C_; ++j) {
            pr[0*C_+j] = e1.x * Us[0][(2*ln)*C_ + j] + e1.y * Us[0][(2*ln+1)*C_ + j];
            pr[1*C_+j] = e2.x * Us[1][(2*ln)*C_ + j] + e2.y * Us[1][(2*ln+1)*C_ + j];
            pr[2*C_+j] = e3.x * Us[2][(2*ln)*C_ + j] + e3.y * Us[2][(2*ln+1)*C_ + j];
        }
        #pragma unroll
        for (int off = 16; off >= 1; off >>= 1) {
            #pragma unroll
            for (int t = 0; t < 3 * C_; ++t)
                pr[t] += __shfl_xor(pr[t], off);
        }
        if (ln == 0) {
            #pragma unroll
            for (int t = 0; t < 3 * C_; ++t) abc[wv][s][t] = pr[t];
        }
    }
    __syncthreads();

    // ---------------- phase 2: WMMA GEMM [16x64]@[64x112] + fused relu/fc dot --
    const int half = ln >> 4;      // K-half of the bf16 A/B layout
    const int col  = ln & 15;      // A row (sample) / B,C,D column

    // A fragments: k(e) = 32*chunk + 8*half + e (+8 if e>=8), contiguous runs of 8
    v16bf aF[2];
    #pragma unroll
    for (int ch = 0; ch < 2; ++ch)
        #pragma unroll
        for (int g = 0; g < 2; ++g) {
            const float* p = &xs[wv][col][ch * 32 + half * 8 + g * 16];
            #pragma unroll
            for (int e = 0; e < 8; ++e)
                aF[ch][g * 8 + e] = (bf16_t)p[e];
        }

    float accg[8];
    #pragma unroll
    for (int r = 0; r < 8; ++r) accg[r] = 0.0f;

    #pragma unroll
    for (int nt = 0; nt < KPAD / 16; ++nt) {
        const int n = nt * 16 + col;
        v16bf bF[2];
        #pragma unroll
        for (int ch = 0; ch < 2; ++ch)
            #pragma unroll
            for (int g = 0; g < 2; ++g) {
                const bf16_t* p = &wT[n][ch * 32 + half * 8 + g * 16];
                #pragma unroll
                for (int e = 0; e < 8; ++e)
                    bF[ch][g * 8 + e] = p[e];
            }
        v8f c8 = {};
        c8 = __builtin_amdgcn_wmma_f32_16x16x32_bf16(false, aF[0], false, bF[0],
                                                     (short)0, c8, false, false);
        c8 = __builtin_amdgcn_wmma_f32_16x16x32_bf16(false, aF[1], false, bF[1],
                                                     (short)0, c8, false, false);
        const float fw = fcg[n];                 // relu + fc dot fused on the tile
        #pragma unroll
        for (int r = 0; r < 8; ++r) {
            float v = c8[r];
            v = v > 0.0f ? v : 0.0f;
            accg[r] += v * fw;
        }
    }
    // reduce over the 16 N-columns within each lane-half (wave32 shuffles)
    #pragma unroll
    for (int off = 1; off < 16; off <<= 1)
        #pragma unroll
        for (int r = 0; r < 8; ++r)
            accg[r] += __shfl_xor(accg[r], off);
    if (col == 0) {
        #pragma unroll
        for (int r = 0; r < 8; ++r)
            gpart[wv][half * 8 + r] = accg[r];   // sample m = r + 8*half
    }
    __syncthreads();

    // ---------------- phase 3: Tucker contraction + epilogue --------------------
    if (ln < SPW && (sbase + ln) < Nn) {
        const float* pr = &abc[wv][ln][0];
        float av[C_], bv[C_], cv[C_];
        #pragma unroll
        for (int j = 0; j < C_; ++j) { av[j] = pr[j]; bv[j] = pr[C_ + j]; cv[j] = pr[2*C_ + j]; }
        float t = 0.0f;
        int q = 0;
        #pragma unroll
        for (int a = 0; a < C_; ++a)
            #pragma unroll
            for (int b = 0; b < C_; ++b) {
                const float ab = av[a] * bv[b];
                #pragma unroll
                for (int c = 0; c < C_; ++c)
                    t += ab * cv[c] * fct[q++];
            }
        out[sbase + ln] = gpart[wv][ln] + t + fcb[0];
    }
}

extern "C" void kernel_launch(void* const* d_in, const int* in_sizes, int n_in,
                              void* d_out, int out_size, void* d_ws, size_t ws_size,
                              hipStream_t stream) {
    const int*   iI  = (const int*)d_in[0];
    const int*   jI  = (const int*)d_in[1];
    const int*   kI  = (const int*)d_in[2];
    const float* uE  = (const float*)d_in[3];
    const float* itE = (const float*)d_in[4];
    const float* tE  = (const float*)d_in[5];
    const float* w   = (const float*)d_in[6];
    const float* U1  = (const float*)d_in[7];
    const float* U2  = (const float*)d_in[8];
    const float* U3  = (const float*)d_in[9];
    const float* fcw = (const float*)d_in[10];
    const float* fcb = (const float*)d_in[11];
    float* out = (float*)d_out;

    const int Nn = in_sizes[0];
    const int blocks = (Nn + SPB - 1) / SPB;
    ntm_fused_kernel<<<blocks, 256, 0, stream>>>(iI, jI, kI, uE, itE, tE, w,
                                                 U1, U2, U3, fcw, fcb, out, Nn);
}
